// DualSymmetryGNN_90993177133723
// MI455X (gfx1250) — matrix operand order repaired and checked by reading, not compiled, not run on hardware
//
#include <hip/hip_runtime.h>

// ---------------- types ----------------
typedef __bf16  bf16x16 __attribute__((ext_vector_type(16)));
typedef float   v8f     __attribute__((ext_vector_type(8)));

__device__ __forceinline__ unsigned short f2bf(float f) {
  unsigned u = __float_as_uint(f);
  unsigned r = 0x7FFFu + ((u >> 16) & 1u);   // round-to-nearest-even
  return (unsigned short)((u + r) >> 16);
}

__device__ __forceinline__ void atomicMaxF(float* addr, float v) {
  if (v >= 0.0f) atomicMax((int*)addr, __float_as_int(v));
  else           atomicMin((unsigned int*)addr, __float_as_uint(v));
}

// ---------------- small utility kernels ----------------
__global__ void fill_val_kernel(float* p, float v, int n) {
  int t = blockIdx.x * 256 + threadIdx.x;
  if (t < n) p[t] = v;
}

__global__ void fill_bias_rows_kernel(float* p, const float* __restrict__ bias,
                                      int rows, int cols) {
  int t = blockIdx.x * 256 + threadIdx.x;
  if (t < rows * cols) p[t] = bias[t & (cols - 1)];   // cols is a power of two
}

__global__ void f32_to_bf16_kernel(const float* __restrict__ in,
                                   unsigned short* __restrict__ out, int n) {
  int t = blockIdx.x * 256 + threadIdx.x;
  if (t < n) out[t] = f2bf(in[t]);
}

// Pack weight W[K,N] (f32) into WMMA B-fragment order (bf16):
// Bp[((nt*(K/32)+kb)*32 + lane)*16 + e] = W[kb*32 + (e>>3)*16 + (lane>>4)*8 + (e&7), nt*16 + (lane&15)]
__global__ void pack_b_kernel(const float* __restrict__ W,
                              unsigned short* __restrict__ Bp, int K, int N) {
  int t = blockIdx.x * 256 + threadIdx.x;
  if (t >= K * N) return;
  int e    = t & 15;
  int lane = (t >> 4) & 31;
  int tile = t >> 9;
  int nkb  = K >> 5;
  int kb   = tile % nkb;
  int nt   = tile / nkb;
  int k    = kb * 32 + ((e >> 3) * 16) + ((lane >> 4) * 8) + (e & 7);
  int col  = nt * 16 + (lane & 15);
  Bp[t] = f2bf(W[(size_t)k * N + col]);
}

// ---------------- WMMA GEMM: C[M,N] = bf16(A[M,K]) @ Bp + bias (+epilogue) ----------------
// Block: 256 threads (8 waves), 2 M-tiles (32 rows) x 8 N-tiles (128 cols).
// Each wave: one B fragment reused across two A fragments -> 2 wmma per global B pair.
#define GF_RELU     1
#define GF_ADD      2
#define GF_RELU_ADD 4

__global__ void gemm_bf16_wmma_kernel(const unsigned short* __restrict__ A,
                                      const unsigned short* __restrict__ Bp,
                                      const float* __restrict__ bias,
                                      const float* __restrict__ addsrc,
                                      float* __restrict__ C,
                                      unsigned short* __restrict__ Cbf,
                                      int M, int K, int N, int flags) {
  __shared__ unsigned short ldsA[32 * 520];           // 32 rows, K<=512, pad stride K+8
  const int tid  = threadIdx.x;
  const int wave = tid >> 5;
  const int lane = tid & 31;
  const int m0   = blockIdx.x * 32;
  const int Kp   = K + 8;

  // cooperative stage of A tile [32,K] -> LDS (16B chunks)
  const int kq = K >> 3;                              // 16B chunks per row
  for (int c = tid; c < 32 * kq; c += 256) {
    int row = c / kq;
    int kc  = (c - row * kq) * 8;
    *(uint4*)&ldsA[row * Kp + kc] =
        *(const uint4*)&A[(size_t)(m0 + row) * K + kc];
  }
  __syncthreads();

  const int nt    = blockIdx.y * 8 + wave;            // global 16-col tile
  const int row16 = lane & 15;
  const int half  = lane >> 4;
  const int nkb   = K >> 5;

  v8f acc0 = {0.f, 0.f, 0.f, 0.f, 0.f, 0.f, 0.f, 0.f};
  v8f acc1 = {0.f, 0.f, 0.f, 0.f, 0.f, 0.f, 0.f, 0.f};
  union Frag { bf16x16 v; uint4 q[2]; };

  // Packed B is contiguous in kb: one k-block = 32 lanes * 32B = 1024B = 64 uint4.
  const uint4* bp = (const uint4*)&Bp[((size_t)nt * nkb * 32 + lane) * 16];
  const unsigned short* la0 = &ldsA[row16 * Kp + half * 8];          // M-tile 0
  const unsigned short* la1 = &ldsA[(16 + row16) * Kp + half * 8];   // M-tile 1

  for (int kb = 0; kb < nkb; kb++) {
    Frag a0, a1, b;
    b.q[0] = bp[0];
    b.q[1] = bp[1];
    __builtin_prefetch(bp + 64, 0, 1);                // next kb's B frag -> global_prefetch
    a0.q[0] = *(const uint4*)(la0);
    a0.q[1] = *(const uint4*)(la0 + 16);
    a1.q[0] = *(const uint4*)(la1);
    a1.q[1] = *(const uint4*)(la1 + 16);
    acc0 = __builtin_amdgcn_wmma_f32_16x16x32_bf16(
        false, a0.v, false, b.v, (short)0, acc0, false, false);
    acc1 = __builtin_amdgcn_wmma_f32_16x16x32_bf16(
        false, a1.v, false, b.v, (short)0, acc1, false, false);
    bp  += 64;                                        // next k-block (1024B)
    la0 += 32;                                        // next 32 K values
    la1 += 32;
  }

  const int col = nt * 16 + (lane & 15);
  const float bv = bias ? bias[col] : 0.f;
#pragma unroll
  for (int t = 0; t < 2; t++) {
    v8f acc = t ? acc1 : acc0;
    const int rowbase = m0 + t * 16 + half * 8;
#pragma unroll
    for (int r = 0; r < 8; r++) {
      size_t o = (size_t)(rowbase + r) * N + col;
      float v = acc[r] + bv;
      if (flags & GF_ADD) {
        float x = addsrc[o];
        if (flags & GF_RELU_ADD) x = fmaxf(x, 0.f);
        v += x;
      }
      if (flags & GF_RELU) v = fmaxf(v, 0.f);
      C[o] = v;
      if (Cbf) Cbf[o] = f2bf(v);
    }
  }
}

// ---------------- edge attention (pass 1): alpha + segment max ----------------
template <int HEADS>
__global__ void edge_att_kernel(const float* __restrict__ xl, const float* __restrict__ xr,
                                const float* __restrict__ We, const float* __restrict__ att,
                                const int* __restrict__ ei, const float* __restrict__ ea,
                                float* __restrict__ alpha, float* __restrict__ amax, int E) {
  constexpr int D    = HEADS * 128;
  constexpr int CPL4 = D / 128;       // float4s per lane
  constexpr int GS   = 32 / HEADS;    // lanes per head group
  int wave = threadIdx.x >> 5, lane = threadIdx.x & 31;
  int e = blockIdx.x * 8 + wave;
  if (e >= E) return;
  int src = ei[e], dst = ei[E + e];
  float wea = ea[e];
  const float4* XL = (const float4*)(xl + (size_t)src * D);
  const float4* XR = (const float4*)(xr + (size_t)dst * D);
  const float4* WE = (const float4*)We;
  const float4* AT = (const float4*)att;
  float p = 0.f;
#pragma unroll
  for (int i = 0; i < CPL4; i++) {
    int idx = lane * CPL4 + i;
    float4 a = XL[idx], b = XR[idx], we = WE[idx], at = AT[idx];
    float m;
    m = a.x + b.x + wea * we.x; m = (m > 0.f) ? m : 0.2f * m; p += m * at.x;
    m = a.y + b.y + wea * we.y; m = (m > 0.f) ? m : 0.2f * m; p += m * at.y;
    m = a.z + b.z + wea * we.z; m = (m > 0.f) ? m : 0.2f * m; p += m * at.z;
    m = a.w + b.w + wea * we.w; m = (m > 0.f) ? m : 0.2f * m; p += m * at.w;
  }
#pragma unroll
  for (int msk = 1; msk < GS; msk <<= 1) p += __shfl_xor(p, msk, 32);
  if ((lane & (GS - 1)) == 0) {
    int h = lane / GS;
    alpha[(size_t)e * HEADS + h] = p;
    atomicMaxF(&amax[(size_t)dst * HEADS + h], p);
  }
}

// ---------------- edge pass 2: exp + segment sum ----------------
template <int HEADS>
__global__ void edge_exp_kernel(const int* __restrict__ ei, float* __restrict__ alpha,
                                const float* __restrict__ amax, float* __restrict__ asum, int E) {
  int idx = blockIdx.x * 256 + threadIdx.x;
  if (idx >= E * HEADS) return;
  int e = idx / HEADS, h = idx - e * HEADS;
  int dst = ei[E + e];
  float ex = __expf(alpha[idx] - amax[(size_t)dst * HEADS + h]);
  alpha[idx] = ex;
  atomicAdd(&asum[(size_t)dst * HEADS + h], ex);
}

// ---------------- edge pass 3: weighted scatter-add ----------------
template <int HEADS>
__global__ void edge_agg_kernel(const float* __restrict__ xl, const float* __restrict__ alpha,
                                const float* __restrict__ asum, const int* __restrict__ ei,
                                float* __restrict__ out, int E) {
  constexpr int D    = HEADS * 128;
  constexpr int CPL4 = D / 128;
  constexpr int GS   = 32 / HEADS;
  int wave = threadIdx.x >> 5, lane = threadIdx.x & 31;
  int e = blockIdx.x * 8 + wave;
  if (e >= E) return;
  int src = ei[e], dst = ei[E + e];
  int h = lane / GS;
  float w = alpha[(size_t)e * HEADS + h] / asum[(size_t)dst * HEADS + h];
  const float4* XL = (const float4*)(xl + (size_t)src * D);
  float* O = out + (size_t)dst * D;
#pragma unroll
  for (int i = 0; i < CPL4; i++) {
    int idx = lane * CPL4 + i;
    float4 a = XL[idx];
    atomicAdd(&O[idx * 4 + 0], a.x * w);
    atomicAdd(&O[idx * 4 + 1], a.y * w);
    atomicAdd(&O[idx * 4 + 2], a.z * w);
    atomicAdd(&O[idx * 4 + 3], a.w * w);
  }
}

// ---------------- fused ReLU + LayerNorm over 512 cols (wave per row) ----------------
__global__ void relu_ln_kernel(const float* __restrict__ in, const float* __restrict__ g,
                               const float* __restrict__ b, float* __restrict__ out,
                               unsigned short* __restrict__ outbf, int rows) {
  int wave = threadIdx.x >> 5, lane = threadIdx.x & 31;
  int row = blockIdx.x * 8 + wave;
  if (row >= rows) return;
  const float4* I = (const float4*)(in + (size_t)row * 512);
  float4 v[4];
  float s = 0.f, sq = 0.f;
#pragma unroll
  for (int i = 0; i < 4; i++) {
    float4 t = I[lane * 4 + i];
    t.x = fmaxf(t.x, 0.f); t.y = fmaxf(t.y, 0.f);
    t.z = fmaxf(t.z, 0.f); t.w = fmaxf(t.w, 0.f);
    v[i] = t;
    s  += t.x + t.y + t.z + t.w;
    sq += t.x * t.x + t.y * t.y + t.z * t.z + t.w * t.w;
  }
#pragma unroll
  for (int msk = 1; msk < 32; msk <<= 1) {
    s  += __shfl_xor(s,  msk, 32);
    sq += __shfl_xor(sq, msk, 32);
  }
  float mu   = s * (1.f / 512.f);
  float var  = sq * (1.f / 512.f) - mu * mu;
  float rstd = rsqrtf(var + 1e-5f);
  const float4* G = (const float4*)g;
  const float4* B = (const float4*)b;
  float4* O = (float4*)(out + (size_t)row * 512);
#pragma unroll
  for (int i = 0; i < 4; i++) {
    int idx = lane * 4 + i;
    float4 gg = G[idx], bb = B[idx], t = v[i];
    float4 o;
    o.x = (t.x - mu) * rstd * gg.x + bb.x;
    o.y = (t.y - mu) * rstd * gg.y + bb.y;
    o.z = (t.z - mu) * rstd * gg.z + bb.z;
    o.w = (t.w - mu) * rstd * gg.w + bb.w;
    O[idx] = o;
    if (outbf) {
      unsigned short* ob = outbf + (size_t)row * 512 + idx * 4;
      ob[0] = f2bf(o.x); ob[1] = f2bf(o.y); ob[2] = f2bf(o.z); ob[3] = f2bf(o.w);
    }
  }
}

// ---------------- graph pooling ----------------
__global__ void pool_kernel(const float* __restrict__ x3, const int* __restrict__ batch,
                            float* __restrict__ padd, float* __restrict__ pmax, int Nn) {
  int t = blockIdx.x * 256 + threadIdx.x;
  if (t >= Nn * 128) return;
  int n = t >> 7, col = t & 127;
  int g = batch[n];
  float v = x3[t];
  atomicAdd(&padd[g * 128 + col], v);
  atomicMaxF(&pmax[g * 128 + col], v);
}

__global__ void pool_fin_kernel(const float* __restrict__ padd, const float* __restrict__ pmax,
                                float* __restrict__ xg) {
  int t = blockIdx.x * 256 + threadIdx.x;
  if (t >= 32 * 256) return;
  int g = t >> 8, col = t & 255;
  if (col < 128) {
    xg[t] = padd[g * 128 + col];
  } else {
    float pm = pmax[g * 128 + (col - 128)];
    bool fin = (pm == pm) && (pm > -3.0e38f) && (pm < 3.0e38f);
    xg[t] = fin ? pm : 0.f;
  }
}

// ---------------- head MLPs (single block) ----------------
__global__ void head_kernel(const float* __restrict__ xg,
                            const float* __restrict__ shW, const float* __restrict__ shb,
                            const float* __restrict__ rg1W, const float* __restrict__ rg1b,
                            const float* __restrict__ rg2W, const float* __restrict__ rg2b,
                            const float* __restrict__ bn1W, const float* __restrict__ bn1b,
                            const float* __restrict__ bn2W, const float* __restrict__ bn2b,
                            float* __restrict__ out) {
  __shared__ float sh[32 * 128];
  __shared__ float t1[32 * 64];
  int tid = threadIdx.x;
  for (int o = tid; o < 32 * 128; o += 256) {
    int gi = o >> 7, c = o & 127;
    float s = shb[c];
    for (int k = 0; k < 256; k++) s += xg[gi * 256 + k] * shW[k * 128 + c];
    sh[o] = fmaxf(s, 0.f);
  }
  __syncthreads();
  for (int o = tid; o < 32 * 64; o += 256) {
    int gi = o >> 6, c = o & 63;
    float s = rg1b[c];
    for (int k = 0; k < 128; k++) s += sh[gi * 128 + k] * rg1W[k * 64 + c];
    t1[o] = fmaxf(s, 0.f);
  }
  __syncthreads();
  if (tid < 32) {
    float s = rg2b[0];
    for (int k = 0; k < 64; k++) s += t1[tid * 64 + k] * rg2W[k];
    out[tid] = s;
  }
  __syncthreads();
  for (int o = tid; o < 32 * 64; o += 256) {
    int gi = o >> 6, c = o & 63;
    float s = bn1b[c];
    for (int k = 0; k < 128; k++) s += sh[gi * 128 + k] * bn1W[k * 64 + c];
    t1[o] = fmaxf(s, 0.f);
  }
  __syncthreads();
  if (tid < 64) {
    int gi = tid >> 1, j = tid & 1;
    float s = bn2b[j];
    for (int k = 0; k < 64; k++) s += t1[gi * 64 + k] * bn2W[k * 2 + j];
    out[32 + gi * 2 + j] = s;
  }
}

// ---------------- host orchestration ----------------
extern "C" void kernel_launch(void* const* d_in, const int* in_sizes, int n_in,
                              void* d_out, int out_size, void* d_ws, size_t ws_size,
                              hipStream_t stream) {
  const int N = 20000, E = 320000, FIN = 128, D = 512, HID = 128, G = 32;

  const float* x     = (const float*)d_in[0];
  const int*   ei    = (const int*)d_in[1];
  const float* ea    = (const float*)d_in[2];
  const int*   batch = (const int*)d_in[3];
  // c1
  const float* c1Wl = (const float*)d_in[4];  const float* c1bl = (const float*)d_in[5];
  const float* c1Wr = (const float*)d_in[6];  const float* c1br = (const float*)d_in[7];
  const float* c1We = (const float*)d_in[8];  const float* c1att = (const float*)d_in[9];
  const float* c1bias = (const float*)d_in[10];
  // c2
  const float* c2Wl = (const float*)d_in[11]; const float* c2bl = (const float*)d_in[12];
  const float* c2Wr = (const float*)d_in[13]; const float* c2br = (const float*)d_in[14];
  const float* c2We = (const float*)d_in[15]; const float* c2att = (const float*)d_in[16];
  const float* c2bias = (const float*)d_in[17];
  // c3
  const float* c3Wl = (const float*)d_in[18]; const float* c3bl = (const float*)d_in[19];
  const float* c3Wr = (const float*)d_in[20]; const float* c3br = (const float*)d_in[21];
  const float* c3We = (const float*)d_in[22]; const float* c3att = (const float*)d_in[23];
  const float* c3bias = (const float*)d_in[24];
  const float* n1g = (const float*)d_in[25];  const float* n1b = (const float*)d_in[26];
  const float* n2g = (const float*)d_in[27];  const float* n2b = (const float*)d_in[28];
  const float* r1W = (const float*)d_in[29];  const float* r1b = (const float*)d_in[30];
  const float* r2W = (const float*)d_in[31];  const float* r2b = (const float*)d_in[32];
  const float* shW = (const float*)d_in[33];  const float* shb = (const float*)d_in[34];
  const float* rg1W = (const float*)d_in[35]; const float* rg1b = (const float*)d_in[36];
  const float* rg2W = (const float*)d_in[37]; const float* rg2b = (const float*)d_in[38];
  const float* bn1W = (const float*)d_in[39]; const float* bn1b = (const float*)d_in[40];
  const float* bn2W = (const float*)d_in[41]; const float* bn2b = (const float*)d_in[42];

  char* wp = (char*)d_ws;
  auto alloc = [&](size_t bytes) -> void* {
    void* p = (void*)wp;
    wp += (bytes + 255) & ~(size_t)255;
    return p;
  };

  unsigned short* xbf  = (unsigned short*)alloc((size_t)N * FIN * 2);
  float* xl   = (float*)alloc((size_t)N * D * 4);
  float* xr   = (float*)alloc((size_t)N * D * 4);
  float* gat  = (float*)alloc((size_t)N * D * 4);
  float* x1   = (float*)alloc((size_t)N * D * 4);
  unsigned short* x1bf = (unsigned short*)alloc((size_t)N * D * 2);
  float* x2t  = (float*)alloc((size_t)N * D * 4);
  float* x2   = (float*)alloc((size_t)N * D * 4);
  unsigned short* x2bf = (unsigned short*)alloc((size_t)N * D * 2);
  float* x3   = (float*)alloc((size_t)N * HID * 4);
  float* alpha = (float*)alloc((size_t)E * 4 * 4);
  float* amax  = (float*)alloc((size_t)N * 4 * 4);
  float* asum  = (float*)alloc((size_t)N * 4 * 4);
  unsigned short* wl1p = (unsigned short*)alloc((size_t)FIN * D * 2);
  unsigned short* wr1p = (unsigned short*)alloc((size_t)FIN * D * 2);
  unsigned short* wl2p = (unsigned short*)alloc((size_t)D * D * 2);
  unsigned short* wr2p = (unsigned short*)alloc((size_t)D * D * 2);
  unsigned short* r1p  = (unsigned short*)alloc((size_t)D * D * 2);
  unsigned short* wl3p = (unsigned short*)alloc((size_t)D * HID * 2);
  unsigned short* wr3p = (unsigned short*)alloc((size_t)D * HID * 2);
  unsigned short* r2p  = (unsigned short*)alloc((size_t)D * HID * 2);
  float* padd = (float*)alloc((size_t)G * HID * 4);
  float* pmax = (float*)alloc((size_t)G * HID * 4);
  float* xg   = (float*)alloc((size_t)G * 256 * 4);

  const float NEG_INF = -__builtin_inff();
  dim3 T(256);

  // ---- weight packing (bf16 WMMA B-fragment order) ----
  pack_b_kernel<<<dim3(FIN * D / 256), T, 0, stream>>>(c1Wl, wl1p, FIN, D);
  pack_b_kernel<<<dim3(FIN * D / 256), T, 0, stream>>>(c1Wr, wr1p, FIN, D);
  pack_b_kernel<<<dim3(D * D / 256),   T, 0, stream>>>(c2Wl, wl2p, D, D);
  pack_b_kernel<<<dim3(D * D / 256),   T, 0, stream>>>(c2Wr, wr2p, D, D);
  pack_b_kernel<<<dim3(D * D / 256),   T, 0, stream>>>(r1W,  r1p,  D, D);
  pack_b_kernel<<<dim3(D * HID / 256), T, 0, stream>>>(c3Wl, wl3p, D, HID);
  pack_b_kernel<<<dim3(D * HID / 256), T, 0, stream>>>(c3Wr, wr3p, D, HID);
  pack_b_kernel<<<dim3(D * HID / 256), T, 0, stream>>>(r2W,  r2p,  D, HID);

  f32_to_bf16_kernel<<<dim3((N * FIN + 255) / 256), T, 0, stream>>>(x, xbf, N * FIN);

  // ---- layer 1 ----
  gemm_bf16_wmma_kernel<<<dim3(N / 32, D / 128), T, 0, stream>>>(
      xbf, wl1p, c1bl, nullptr, xl, nullptr, N, FIN, D, 0);
  gemm_bf16_wmma_kernel<<<dim3(N / 32, D / 128), T, 0, stream>>>(
      xbf, wr1p, c1br, nullptr, xr, nullptr, N, FIN, D, 0);
  fill_bias_rows_kernel<<<dim3((N * D + 255) / 256), T, 0, stream>>>(gat, c1bias, N, D);
  fill_val_kernel<<<dim3((N * 4 + 255) / 256), T, 0, stream>>>(amax, NEG_INF, N * 4);
  fill_val_kernel<<<dim3((N * 4 + 255) / 256), T, 0, stream>>>(asum, 0.f, N * 4);
  edge_att_kernel<4><<<dim3((E + 7) / 8), T, 0, stream>>>(xl, xr, c1We, c1att, ei, ea, alpha, amax, E);
  edge_exp_kernel<4><<<dim3((E * 4 + 255) / 256), T, 0, stream>>>(ei, alpha, amax, asum, E);
  edge_agg_kernel<4><<<dim3((E + 7) / 8), T, 0, stream>>>(xl, alpha, asum, ei, gat, E);
  relu_ln_kernel<<<dim3((N + 7) / 8), T, 0, stream>>>(gat, n1g, n1b, x1, x1bf, N);

  // ---- layer 2 ----
  gemm_bf16_wmma_kernel<<<dim3(N / 32, D / 128), T, 0, stream>>>(
      x1bf, wl2p, c2bl, nullptr, xl, nullptr, N, D, D, 0);
  gemm_bf16_wmma_kernel<<<dim3(N / 32, D / 128), T, 0, stream>>>(
      x1bf, wr2p, c2br, nullptr, xr, nullptr, N, D, D, 0);
  fill_bias_rows_kernel<<<dim3((N * D + 255) / 256), T, 0, stream>>>(gat, c2bias, N, D);
  fill_val_kernel<<<dim3((N * 4 + 255) / 256), T, 0, stream>>>(amax, NEG_INF, N * 4);
  fill_val_kernel<<<dim3((N * 4 + 255) / 256), T, 0, stream>>>(asum, 0.f, N * 4);
  edge_att_kernel<4><<<dim3((E + 7) / 8), T, 0, stream>>>(xl, xr, c2We, c2att, ei, ea, alpha, amax, E);
  edge_exp_kernel<4><<<dim3((E * 4 + 255) / 256), T, 0, stream>>>(ei, alpha, amax, asum, E);
  edge_agg_kernel<4><<<dim3((E + 7) / 8), T, 0, stream>>>(xl, alpha, asum, ei, gat, E);
  relu_ln_kernel<<<dim3((N + 7) / 8), T, 0, stream>>>(gat, n2g, n2b, x2t, nullptr, N);
  // x2 = LN(relu(gat2)) + x1 @ r1W + r1b
  gemm_bf16_wmma_kernel<<<dim3(N / 32, D / 128), T, 0, stream>>>(
      x1bf, r1p, r1b, x2t, x2, x2bf, N, D, D, GF_ADD);

  // ---- layer 3 (1 head) ----
  gemm_bf16_wmma_kernel<<<dim3(N / 32, HID / 128), T, 0, stream>>>(
      x2bf, wl3p, c3bl, nullptr, xl, nullptr, N, D, HID, 0);
  gemm_bf16_wmma_kernel<<<dim3(N / 32, HID / 128), T, 0, stream>>>(
      x2bf, wr3p, c3br, nullptr, xr, nullptr, N, D, HID, 0);
  fill_bias_rows_kernel<<<dim3((N * HID + 255) / 256), T, 0, stream>>>(gat, c3bias, N, HID);
  fill_val_kernel<<<dim3((N + 255) / 256), T, 0, stream>>>(amax, NEG_INF, N);
  fill_val_kernel<<<dim3((N + 255) / 256), T, 0, stream>>>(asum, 0.f, N);
  edge_att_kernel<1><<<dim3((E + 7) / 8), T, 0, stream>>>(xl, xr, c3We, c3att, ei, ea, alpha, amax, E);
  edge_exp_kernel<1><<<dim3((E + 255) / 256), T, 0, stream>>>(ei, alpha, amax, asum, E);
  edge_agg_kernel<1><<<dim3((E + 7) / 8), T, 0, stream>>>(xl, alpha, asum, ei, gat, E);
  // x3 = relu(gat3) + x2 @ r2W + r2b
  gemm_bf16_wmma_kernel<<<dim3(N / 32, HID / 128), T, 0, stream>>>(
      x2bf, r2p, r2b, gat, x3, nullptr, N, D, HID, GF_ADD | GF_RELU_ADD);

  // ---- pooling + head ----
  fill_val_kernel<<<dim3((G * HID + 255) / 256), T, 0, stream>>>(padd, 0.f, G * HID);
  fill_val_kernel<<<dim3((G * HID + 255) / 256), T, 0, stream>>>(pmax, NEG_INF, G * HID);
  pool_kernel<<<dim3((N * HID + 255) / 256), T, 0, stream>>>(x3, batch, padd, pmax, N);
  pool_fin_kernel<<<dim3((G * 256 + 255) / 256), T, 0, stream>>>(padd, pmax, xg);
  head_kernel<<<dim3(1), T, 0, stream>>>(xg, shW, shb, rg1W, rg1b, rg2W, rg2b,
                                         bn1W, bn1b, bn2W, bn2b, (float*)d_out);
}
